// TorchKNNImputer_88459146428492
// MI455X (gfx1250) — matrix operand order repaired and checked by reading, not compiled, not run on hardware
//
#include <hip/hip_runtime.h>
#include <hip/hip_bf16.h>
#include <stdint.h>

// Problem constants (match reference)
#define NQ   16384
#define NF   8192
#define DIM  64
#define KNN  5
#define KSEG 192          // combined K: [XX|oX|-2Xc] . [oY|YY|Yc]
#define KCH  6            // KSEG / 32
#define QCHUNK 2048       // query rows per chunk (dist chunk = 64 MB, L2-resident)

typedef _Float16 h16;
typedef __attribute__((ext_vector_type(16))) _Float16 v16h;
typedef __attribute__((ext_vector_type(8)))  float    v8f;

// ---- WMMA 16-bit fragment lane layouts (ISA 7.12.2) ----
// A 16x32: lanes0-15 M=lane, V0..3 hold K=0..7 (2/VGPR), V4..7 hold K=16..23;
//          lanes16-31 same M, K offset +8.
__device__ __forceinline__ int a_klocal(int lane, int e) {
    int v = e >> 1, h = e & 1;
    int k = (v < 4) ? (v * 2 + h) : (16 + (v - 4) * 2 + h);
    if (lane >= 16) k += 8;
    return k;
}
// B 32x16: lanes0-15 N=lane, V0..3 hold K=0..7, V4..7 hold K=16..23;
//          lanes16-31 same N, K offset +8 (matches A).
__device__ __forceinline__ int b_klocal(int lane, int e) {
    int v = e >> 1, h = e & 1;
    int k = (v < 4) ? (v * 2 + h) : (16 + (v - 4) * 2 + h);
    if (lane >= 16) k += 8;
    return k;
}

// ---- Build A operand in fragment order: segments [Xc^2 | obsX | -2*Xc] ----
__global__ void prepA(const float* __restrict__ X, h16* __restrict__ Aq) {
    int tid = blockIdx.x * blockDim.x + threadIdx.x;      // one thread = one lane-fragment (16 halves)
    const int perTile = KCH * 32;
    if (tid >= (NQ / 16) * perTile) return;
    int mt   = tid / perTile;
    int r    = tid % perTile;
    int kc   = r / 32;
    int lane = r % 32;
    int row  = mt * 16 + (lane & 15);
    h16* dst = Aq + ((size_t)(mt * KCH + kc) * 32 + lane) * 16;
    #pragma unroll
    for (int e = 0; e < 16; ++e) {
        int gk  = kc * 32 + a_klocal(lane, e);
        int seg = gk >> 6, j = gk & 63;
        float x  = X[(size_t)row * DIM + j];
        bool miss = (x != x);
        float xc  = miss ? 0.f : x;
        float val = (seg == 0) ? xc * xc
                  : (seg == 1) ? (miss ? 0.f : 1.f)
                  : -2.f * xc;
        dst[e] = (h16)val;
    }
}

// ---- Build B operand in fragment order: segments [obsY | Yc^2 | Yc] ----
__global__ void prepB(const float* __restrict__ F, h16* __restrict__ Bf) {
    int tid = blockIdx.x * blockDim.x + threadIdx.x;
    const int perTile = KCH * 32;
    if (tid >= (NF / 16) * perTile) return;
    int nt   = tid / perTile;
    int r    = tid % perTile;
    int kc   = r / 32;
    int lane = r % 32;
    int row  = nt * 16 + (lane & 15);
    h16* dst = Bf + ((size_t)(nt * KCH + kc) * 32 + lane) * 16;
    #pragma unroll
    for (int e = 0; e < 16; ++e) {
        int gk  = kc * 32 + b_klocal(lane, e);
        int seg = gk >> 6, j = gk & 63;
        float y  = F[(size_t)row * DIM + j];
        bool miss = (y != y);
        float yc  = miss ? 0.f : y;
        float val = (seg == 0) ? (miss ? 0.f : 1.f)
                  : (seg == 1) ? yc * yc
                  : yc;
        dst[e] = (h16)val;
    }
}

// ---- Observed-feature bitmask per fit row ----
__global__ void maskk(const float* __restrict__ F, uint64_t* __restrict__ m) {
    int f = blockIdx.x * blockDim.x + threadIdx.x;
    if (f >= NF) return;
    uint64_t b = 0;
    #pragma unroll
    for (int c = 0; c < DIM; ++c) {
        float v = F[(size_t)f * DIM + c];
        if (!(v != v)) b |= (1ull << c);
    }
    m[f] = b;
}

// ---- Column means of observed fit values (fallback) ----
__global__ void colmeank(const float* __restrict__ F, float* __restrict__ cm) {
    int c = blockIdx.x, t = threadIdx.x;
    __shared__ float ssum[256];
    __shared__ int   scnt[256];
    float s = 0.f; int n = 0;
    for (int f = t; f < NF; f += 256) {
        float v = F[(size_t)f * DIM + c];
        if (!(v != v)) { s += v; n++; }
    }
    ssum[t] = s; scnt[t] = n;
    __syncthreads();
    for (int st = 128; st > 0; st >>= 1) {
        if (t < st) { ssum[t] += ssum[t + st]; scnt[t] += scnt[t + st]; }
        __syncthreads();
    }
    if (t == 0) cm[c] = (scnt[0] > 0) ? ssum[0] / (float)scnt[0] : 0.f;
}

// ---- WMMA GEMM: ranking keys key[q,f] = max(d,0)/present (inf if present==0) ----
// Block = 8 waves; wave computes 16(M) x 64(N). The block's 24 KB B tile is
// staged once into LDS with CDNA5 async global->LDS copies (ASYNCcnt), then
// all 8 waves feed WMMA from LDS (8x reduction in L2 traffic for B).
__global__ void __launch_bounds__(256) gemmk(const h16* __restrict__ Aq,
                                             const h16* __restrict__ Bf,
                                             float* __restrict__ distc,
                                             int qbase) {
    __shared__ h16 sB[4 * KCH * 32 * 16];       // 24 KB: 4 N-tiles x 6 K-chunks

    int lane  = threadIdx.x & 31;
    int wave  = threadIdx.x >> 5;
    int mtLoc = blockIdx.x * 8 + wave;          // tile row within chunk (0..127)
    int mtG   = (qbase >> 4) + mtLoc;           // global A tile
    int nt0   = blockIdx.y * 4;

    // --- async stage of B tile (contiguous 24 KB region) into LDS ---
    {
        uint32_t ldsbase = (uint32_t)(uintptr_t)(&sB[0]);
        uint64_t gbase   = (uint64_t)(uintptr_t)(Bf + (size_t)nt0 * KCH * 32 * 16);
        #pragma unroll
        for (int i = 0; i < 6; ++i) {
            uint32_t o = (uint32_t)((i * 256 + threadIdx.x) * 16);
            asm volatile("global_load_async_to_lds_b128 %0, %1, off"
                         :: "v"(ldsbase + o), "v"(gbase + (uint64_t)o)
                         : "memory");
        }
        asm volatile("s_wait_asynccnt 0" ::: "memory");
    }

    // A fragments (reused across 4 N-tiles, loaded after the async-copy wait)
    v16h a[KCH];
    #pragma unroll
    for (int kc = 0; kc < KCH; ++kc)
        a[kc] = *(const v16h*)(Aq + ((size_t)(mtG * KCH + kc) * 32 + lane) * 16);

    __syncthreads();                            // B tile visible to all waves

    #pragma unroll
    for (int n = 0; n < 4; ++n) {
        v8f accd = {0.f,0.f,0.f,0.f,0.f,0.f,0.f,0.f};
        v8f accp = {0.f,0.f,0.f,0.f,0.f,0.f,0.f,0.f};
        #pragma unroll
        for (int kc = 0; kc < KCH; ++kc) {
            v16h b = *(const v16h*)(&sB[((n * KCH + kc) * 32 + lane) * 16]);
            // d accumulation: segment-aligned A.B over all K=192
            accd = __builtin_amdgcn_wmma_f32_16x16x32_f16(
                       false, a[kc], false, b, (short)0, accd, false, false);
            // present = oX . oY : A segment1 (kc=2,3) x B segment0 (kc=0,1)
            if (kc < 2)
                accp = __builtin_amdgcn_wmma_f32_16x16x32_f16(
                           false, a[kc + 2], false, b, (short)0, accp, false, false);
        }
        // C/D layout: VGPR r -> M = r (+8 for lanes 16-31); N = lane&15
        int col     = (nt0 + n) * 16 + (lane & 15);
        int rowBase = mtLoc * 16 + ((lane >= 16) ? 8 : 0);
        #pragma unroll
        for (int r = 0; r < 8; ++r) {
            float p  = accp[r];
            float dd = accd[r];
            // ranking key only: fast reciprocal (p is an exact integer 1..64),
            // branchless select; present==0 -> +inf (excluded in selection)
            float key = fmaxf(dd, 0.f) * __builtin_amdgcn_rcpf(p);
            key = (p > 0.5f) ? key : __builtin_inff();
            distc[(size_t)(rowBase + r) * NF + col] = key;
        }
    }
}

// ---- Per-row imputation: per missing column, top-5 valid donors, uniform mean ----
__global__ void __launch_bounds__(256) imputek(const float* __restrict__ distc,
                                               const uint64_t* __restrict__ obsm,
                                               const float* __restrict__ cm,
                                               const float* __restrict__ X,
                                               const float* __restrict__ F,
                                               float* __restrict__ out,
                                               int qbase) {
    int q = qbase + blockIdx.x;
    int t = threadIdx.x;
    __shared__ float sX[DIM];
    __shared__ float sres[DIM];
    __shared__ float skey[256 * KNN];
    __shared__ int   sidx[256 * KNN];
    __shared__ float skey2[32 * KNN];
    __shared__ int   sidx2[32 * KNN];

    if (t < DIM) { sX[t] = X[(size_t)q * DIM + t]; sres[t] = 0.f; }
    __syncthreads();

    const float* drow = distc + (size_t)blockIdx.x * NF;
    const float INF = __builtin_inff();

    for (int c = 0; c < DIM; ++c) {
        float xv = sX[c];
        if (!(xv != xv)) continue;              // uniform across block

        // stage 1: per-thread top-5 over strided donors
        float bk[KNN]; int bi[KNN];
        #pragma unroll
        for (int i = 0; i < KNN; ++i) { bk[i] = INF; bi[i] = -1; }
        for (int f = t; f < NF; f += 256) {
            if (((obsm[f] >> c) & 1ull) == 0) continue;
            float k = drow[f];
            if (k >= bk[KNN - 1]) continue;     // +inf keys (present==0) never enter
            int p = KNN - 1;
            while (p > 0 && bk[p - 1] > k) { bk[p] = bk[p - 1]; bi[p] = bi[p - 1]; --p; }
            bk[p] = k; bi[p] = f;
        }
        #pragma unroll
        for (int i = 0; i < KNN; ++i) { skey[t * KNN + i] = bk[i]; sidx[t * KNN + i] = bi[i]; }
        __syncthreads();

        // stage 2: 32 threads merge 1280 -> 160
        if (t < 32) {
            float ck[KNN]; int ci[KNN];
            #pragma unroll
            for (int i = 0; i < KNN; ++i) { ck[i] = INF; ci[i] = -1; }
            for (int j = t * 40; j < t * 40 + 40; ++j) {
                float k = skey[j];
                if (k >= ck[KNN - 1]) continue;
                int id = sidx[j];
                int p = KNN - 1;
                while (p > 0 && ck[p - 1] > k) { ck[p] = ck[p - 1]; ci[p] = ci[p - 1]; --p; }
                ck[p] = k; ci[p] = id;
            }
            #pragma unroll
            for (int i = 0; i < KNN; ++i) { skey2[t * KNN + i] = ck[i]; sidx2[t * KNN + i] = ci[i]; }
        }
        __syncthreads();

        // stage 3: final top-5 over 160 + uniform-weight mean (fallback: column mean)
        if (t == 0) {
            float fk[KNN]; int fi[KNN];
            #pragma unroll
            for (int i = 0; i < KNN; ++i) { fk[i] = INF; fi[i] = -1; }
            for (int j = 0; j < 32 * KNN; ++j) {
                float k = skey2[j];
                if (k >= fk[KNN - 1]) continue;
                int id = sidx2[j];
                int p = KNN - 1;
                while (p > 0 && fk[p - 1] > k) { fk[p] = fk[p - 1]; fi[p] = fi[p - 1]; --p; }
                fk[p] = k; fi[p] = id;
            }
            float sum = 0.f; int cnt = 0;
            #pragma unroll
            for (int i = 0; i < KNN; ++i) {
                if (fi[i] >= 0 && fk[i] < INF) { sum += F[(size_t)fi[i] * DIM + c]; cnt++; }
            }
            sres[c] = (cnt > 0) ? sum / (float)cnt : cm[c];
        }
        __syncthreads();
    }

    if (t < DIM) {
        float xv = sX[t];
        out[(size_t)q * DIM + t] = (xv != xv) ? sres[t] : xv;
    }
}

extern "C" void kernel_launch(void* const* d_in, const int* in_sizes, int n_in,
                              void* d_out, int out_size, void* d_ws, size_t ws_size,
                              hipStream_t stream) {
    (void)in_sizes; (void)n_in; (void)out_size; (void)ws_size;
    const float* X = (const float*)d_in[0];      // [NQ, 64]
    const float* F = (const float*)d_in[1];      // [NF, 64]
    // d_in[2] = n_neighbors (==5 per reference; KNN is compile-time)
    float* out = (float*)d_out;

    // workspace layout (all offsets 256B-aligned by construction)
    char* ws = (char*)d_ws;
    float*    distc = (float*)ws;                                      // QCHUNK*NF*4 = 64 MB
    h16*      Aq    = (h16*)(ws + (size_t)QCHUNK * NF * 4);            // NQ*192*2  = 6 MB
    h16*      Bf    = (h16*)((char*)Aq + (size_t)NQ * KSEG * 2);       // NF*192*2  = 3 MB
    uint64_t* msk   = (uint64_t*)((char*)Bf + (size_t)NF * KSEG * 2);  // NF*8      = 64 KB
    float*    cm    = (float*)((char*)msk + (size_t)NF * 8);           // 64 floats

    {
        int total = (NQ / 16) * KCH * 32;
        prepA<<<(total + 255) / 256, 256, 0, stream>>>(X, Aq);
    }
    {
        int total = (NF / 16) * KCH * 32;
        prepB<<<(total + 255) / 256, 256, 0, stream>>>(F, Bf);
    }
    maskk<<<NF / 256, 256, 0, stream>>>(F, msk);
    colmeank<<<DIM, 256, 0, stream>>>(F, cm);

    for (int qb = 0; qb < NQ; qb += QCHUNK) {
        gemmk<<<dim3(QCHUNK / 128, NF / 64), 256, 0, stream>>>(Aq, Bf, distc, qb);
        imputek<<<QCHUNK, 256, 0, stream>>>(distc, msk, cm, X, F, out, qb);
    }
}
